// RegularizedFMNet_77438260347209
// MI455X (gfx1250) — matrix operand order repaired and checked by reading, not compiled, not run on hardware
//
#include <hip/hip_runtime.h>
#include <stdint.h>

// Problem constants (from reference setup_inputs)
#define NB 8
#define NDIM 20000
#define KD 128
#define CD 256

// GEMM tiling
#define SPLITK 8
#define CHUNK 2500   // NDIM / SPLITK
#define KB 64        // k per staged step (2 WMMA k-substeps of 32)
#define TC 64        // output cols per workgroup

#define NTRI ((KD * (KD + 1)) / 2)

typedef __attribute__((ext_vector_type(16))) __bf16        bf16x16;
typedef __attribute__((ext_vector_type(8)))  float         f32x8;
typedef __attribute__((ext_vector_type(4)))  unsigned int  u32x4;

union Frag { bf16x16 v; u32x4 q[2]; };

// Split f32 into bf16 hi + bf16 lo (truncation); hi*hi + hi*lo + lo*hi ~ fp32.
__device__ __forceinline__ void split_bf16(float x, unsigned short& hi, unsigned short& lo) {
    unsigned int ux = __float_as_uint(x);
    hi = (unsigned short)(ux >> 16);
    float fhi = __uint_as_float(ux & 0xffff0000u);
    lo = (unsigned short)(__float_as_uint(x - fhi) >> 16);
}

// ---------------------------------------------------------------------------
// Kernel 1: split-K projection GEMM  part[s,b,:,:] = E[b,:,kchunk] * F[b,kchunk,:]
// E: [NB, KD, NDIM]  (evecs_trans), F: [NB, NDIM, CD] (feat)
// part: [SPLITK, NB, KD, CD] f32
// Register double-buffered: global loads for step st+1 overlap WMMAs of step st.
// ---------------------------------------------------------------------------
__global__ __launch_bounds__(256) void proj_gemm_kernel(
    const float* __restrict__ E, const float* __restrict__ F, float* __restrict__ part)
{
    __shared__ unsigned short sEhi[KD * KB];   // 16 KB
    __shared__ unsigned short sElo[KD * KB];   // 16 KB
    __shared__ unsigned short sFhi[TC * KB];   //  8 KB, transposed: [col][k]
    __shared__ unsigned short sFlo[TC * KB];   //  8 KB

    const int s    = blockIdx.x;
    const int c0   = blockIdx.y * TC;
    const int b    = blockIdx.z;
    const int tid  = threadIdx.x;
    const int wave = tid >> 5;
    const int lane = tid & 31;
    const int lh   = lane & 15;
    const int sel  = lane >> 4;

    const float* Eb = E + (size_t)b * KD * NDIM;
    const float* Fb = F + (size_t)b * NDIM * CD;

    f32x8 acc[4];
    for (int t = 0; t < 4; ++t)
        for (int e = 0; e < 8; ++e) acc[t][e] = 0.0f;

    const int kBase  = s * CHUNK;
    const int nSteps = (CHUNK + KB - 1) / KB;   // 40

    // Per-thread staging registers (double buffer against LDS)
    float regE[(KD * KB) / 256];   // 32
    float regF[(KB * TC) / 256];   // 16

    const int rE = tid >> 6, kE = tid & 63;     // E: 4 rows per i-iter
    const int kF = tid >> 6, cF = tid & 63;     // F: 4 k-rows per i-iter

    // Prologue: load step 0
#pragma unroll
    for (int i = 0; i < (KD * KB) / 256; ++i) {
        int r = rE + i * 4;
        regE[i] = (kE < CHUNK) ? Eb[(size_t)r * NDIM + (kBase + kE)] : 0.0f;
    }
#pragma unroll
    for (int i = 0; i < (KB * TC) / 256; ++i) {
        int kk = kF + i * 4;
        regF[i] = (kk < CHUNK) ? Fb[(size_t)(kBase + kk) * CD + (c0 + cF)] : 0.0f;
    }

    for (int st = 0; st < nSteps; ++st) {
        // Convert staged registers -> LDS (bf16 hi/lo)
#pragma unroll
        for (int i = 0; i < (KD * KB) / 256; ++i) {
            int r = rE + i * 4;
            unsigned short h, l; split_bf16(regE[i], h, l);
            sEhi[r * KB + kE] = h; sElo[r * KB + kE] = l;
        }
#pragma unroll
        for (int i = 0; i < (KB * TC) / 256; ++i) {
            int kk = kF + i * 4;
            unsigned short h, l; split_bf16(regF[i], h, l);
            sFhi[cF * KB + kk] = h; sFlo[cF * KB + kk] = l;
        }
        __syncthreads();

        // Issue global loads for next step; they complete during the WMMAs below.
        if (st + 1 < nSteps) {
            const int kOff = (st + 1) * KB;
#pragma unroll
            for (int i = 0; i < (KD * KB) / 256; ++i) {
                int r = rE + i * 4;
                regE[i] = ((kOff + kE) < CHUNK) ? Eb[(size_t)r * NDIM + (kBase + kOff + kE)] : 0.0f;
            }
#pragma unroll
            for (int i = 0; i < (KB * TC) / 256; ++i) {
                int kk = kOff + kF + i * 4;
                regF[i] = (kk < CHUNK) ? Fb[(size_t)(kBase + kk) * CD + (c0 + cF)] : 0.0f;
            }
        }

        // Two WMMA k-substeps of 32
        const int row = wave * 16 + lh;
#pragma unroll
        for (int ks = 0; ks < 2; ++ks) {
            const int eo = row * KB + ks * 32;
            Frag ahi, alo;
            ahi.q[0] = *(const u32x4*)&sEhi[eo + sel * 8];
            ahi.q[1] = *(const u32x4*)&sEhi[eo + 16 + sel * 8];
            alo.q[0] = *(const u32x4*)&sElo[eo + sel * 8];
            alo.q[1] = *(const u32x4*)&sElo[eo + 16 + sel * 8];

#pragma unroll
            for (int ct = 0; ct < 4; ++ct) {
                const int fo = (ct * 16 + lh) * KB + ks * 32;
                Frag bhi, blo;
                bhi.q[0] = *(const u32x4*)&sFhi[fo + sel * 8];
                bhi.q[1] = *(const u32x4*)&sFhi[fo + 16 + sel * 8];
                blo.q[0] = *(const u32x4*)&sFlo[fo + sel * 8];
                blo.q[1] = *(const u32x4*)&sFlo[fo + 16 + sel * 8];

                acc[ct] = __builtin_amdgcn_wmma_f32_16x16x32_bf16(false, ahi.v, false, bhi.v,
                                                                  (short)0, acc[ct], false, false);
                acc[ct] = __builtin_amdgcn_wmma_f32_16x16x32_bf16(false, ahi.v, false, blo.v,
                                                                  (short)0, acc[ct], false, false);
                acc[ct] = __builtin_amdgcn_wmma_f32_16x16x32_bf16(false, alo.v, false, bhi.v,
                                                                  (short)0, acc[ct], false, false);
            }
        }
        __syncthreads();
    }

    // C/D layout: VGPR r -> M = r + 8*sel; lane%16 -> N
    float* pb = part + ((size_t)s * NB + b) * (KD * CD);
#pragma unroll
    for (int ct = 0; ct < 4; ++ct)
#pragma unroll
        for (int r = 0; r < 8; ++r) {
            int m = wave * 16 + sel * 8 + r;
            int c = c0 + ct * 16 + lh;
            pb[(size_t)m * CD + c] = acc[ct][r];
        }
}

// ---------------------------------------------------------------------------
// Kernel 2: deterministic split-K reduction
// ---------------------------------------------------------------------------
__global__ __launch_bounds__(256) void reduce_kernel(
    const float* __restrict__ part, float* __restrict__ out, int n)
{
    int i = blockIdx.x * 256 + threadIdx.x;
    if (i >= n) return;
    float s = 0.0f;
#pragma unroll
    for (int k = 0; k < SPLITK; ++k) s += part[(size_t)k * n + i];
    out[i] = s;
}

// ---------------------------------------------------------------------------
// Kernel 3: Gram matrices  AAt[b,i,j] = sum_c A[b,i,c]A[b,j,c]; BAt uses Bm rows.
// ---------------------------------------------------------------------------
__global__ __launch_bounds__(256) void gram_kernel(
    const float* __restrict__ A, const float* __restrict__ Bm,
    float* __restrict__ AAt, float* __restrict__ BAt)
{
    __shared__ float sp[32][33];
    __shared__ float sq[32][33];
    const int ti = blockIdx.x, tj = blockIdx.y;
    const int b = blockIdx.z >> 1, which = blockIdx.z & 1;
    const int tid = threadIdx.x;
    const float* P = (which ? Bm : A) + (size_t)b * KD * CD;
    const float* Q = A + (size_t)b * KD * CD;
    float* O = (which ? BAt : AAt) + (size_t)b * KD * KD;

    const int jj  = tid & 31;
    const int ii0 = (tid >> 5) * 4;
    float acc[4] = {0.f, 0.f, 0.f, 0.f};

    for (int cc = 0; cc < CD; cc += 32) {
#pragma unroll
        for (int e = 0; e < 4; ++e) {
            int idx = tid + e * 256;
            int r = idx >> 5, c = idx & 31;
            sp[r][c] = P[(size_t)(ti * 32 + r) * CD + cc + c];
            sq[r][c] = Q[(size_t)(tj * 32 + r) * CD + cc + c];
        }
        __syncthreads();
#pragma unroll 8
        for (int c = 0; c < 32; ++c) {
            float qv = sq[jj][c];
#pragma unroll
            for (int e = 0; e < 4; ++e) acc[e] += sp[ii0 + e][c] * qv;
        }
        __syncthreads();
    }
#pragma unroll
    for (int e = 0; e < 4; ++e)
        O[(size_t)(ti * 32 + ii0 + e) * KD + (tj * 32 + jj)] = acc[e];
}

// ---------------------------------------------------------------------------
// Kernel 4: mask D[b,i,j] from evals
// ---------------------------------------------------------------------------
__device__ __forceinline__ float safe_ev(float v) {
    if (v != v) v = 1e-10f;      // nan
    if (v > 1e6f) v = 1e6f;      // +inf
    if (v < 1e-10f) v = 1e-10f;  // -inf / clip
    return v;
}

__global__ __launch_bounds__(256) void mask_kernel(
    const float* __restrict__ ex, const float* __restrict__ ey, float* __restrict__ Dm)
{
    __shared__ float r1[KD], q1[KD], r2[KD], q2[KD];
    __shared__ float red[256];
    const int b = blockIdx.x;
    const int tid = threadIdx.x;

    float e1 = 0.0f, e2 = 0.0f;
    if (tid < KD) {
        e1 = safe_ev(ex[b * KD + tid]);
        e2 = safe_ev(ey[b * KD + tid]);
    }
    red[tid] = fmaxf(e1, e2);
    __syncthreads();
    for (int off = 128; off > 0; off >>= 1) {
        if (tid < off) red[tid] = fmaxf(red[tid], red[tid + off]);
        __syncthreads();
    }
    float scale = fmaxf(red[0], 1e-10f);
    if (tid < KD) {
        float g1 = sqrtf(e1 / scale); float d1 = g1 * g1 + 1.0f;  // gamma = 0.5
        r1[tid] = g1 / d1; q1[tid] = 1.0f / d1;
        float g2 = sqrtf(e2 / scale); float d2 = g2 * g2 + 1.0f;
        r2[tid] = g2 / d2; q2[tid] = 1.0f / d2;
    }
    __syncthreads();
    float* Db = Dm + (size_t)b * KD * KD;
    for (int idx = tid; idx < KD * KD; idx += 256) {
        int i = idx >> 7, j = idx & 127;
        float re = r2[i] - r1[j];
        float im = q2[i] - q1[j];
        float v = re * re + im * im;
        if (v != v) v = 0.0f;
        if (v > 1e6f) v = 1e6f;
        if (v < 0.0f) v = 0.0f;
        Db[idx] = v;
    }
}

// ---------------------------------------------------------------------------
// Kernel 5: 1024 SPD solves via Cholesky, packed lower triangle in LDS.
// Division-free trailing update via LDS row-lookup tables.
// grid: (KD, NB); block 256 threads solves system (b, i)
// ---------------------------------------------------------------------------
#define TRI(r, c) ((((r) * ((r) + 1)) >> 1) + (c))

__global__ __launch_bounds__(256) void solve_kernel(
    const float* __restrict__ AAt, const float* __restrict__ BAt,
    const float* __restrict__ Dm, float* __restrict__ out)
{
    __shared__ float Ml[NTRI];            // 33 KB packed lower triangle
    __shared__ float yv[KD];
    __shared__ float pinv;
    __shared__ unsigned char trow[NTRI];  // 8.25 KB: row index of packed slot
    __shared__ int rbase[KD];             // row start offsets

    const int i = blockIdx.x;
    const int b = blockIdx.y;
    const int tid = threadIdx.x;

    const float* G = AAt + (size_t)b * KD * KD;
    const size_t rowOff = ((size_t)b * KD + i) * KD;

    if (tid < KD) rbase[tid] = (tid * (tid + 1)) / 2;

    // Load lower triangle of G (symmetric); fill row table
    for (int t = tid; t < NTRI; t += 256) {
        int r = (int)((sqrtf(8.0f * (float)t + 1.0f) - 1.0f) * 0.5f);
        while (((r + 1) * (r + 2)) / 2 <= t) ++r;
        while ((r * (r + 1)) / 2 > t) --r;
        int c = t - (r * (r + 1)) / 2;
        trow[t] = (unsigned char)r;
        Ml[t] = G[(size_t)r * KD + c];
    }
    if (tid < KD) yv[tid] = BAt[rowOff + tid];
    __syncthreads();
    if (tid < KD) Ml[TRI(tid, tid)] += 100.0f * Dm[rowOff + tid] + 1e-8f;  // LMBDA = 100
    __syncthreads();

    // Cholesky: M = L L^T
    for (int k = 0; k < KD; ++k) {
        if (tid == 0) {
            float d = sqrtf(fmaxf(Ml[TRI(k, k)], 1e-20f));
            Ml[TRI(k, k)] = d;
            pinv = 1.0f / d;
        }
        __syncthreads();
        float pv = pinv;
        for (int r = k + 1 + tid; r < KD; r += 256) Ml[rbase[r] + k] *= pv;
        __syncthreads();
        // Trailing update: scan packed rows k+1..127; update entries with c > k
        const int start = (k + 1 < KD) ? rbase[k + 1] : NTRI;
        for (int t = start + tid; t < NTRI; t += 256) {
            int r = trow[t];
            int c = t - rbase[r];
            if (c > k) Ml[t] -= Ml[rbase[r] + k] * Ml[rbase[c] + k];
        }
        __syncthreads();
    }
    // Forward solve L z = y
    for (int k = 0; k < KD; ++k) {
        if (tid == 0) yv[k] /= Ml[TRI(k, k)];
        __syncthreads();
        float yk = yv[k];
        for (int r = k + 1 + tid; r < KD; r += 256) yv[r] -= Ml[rbase[r] + k] * yk;
        __syncthreads();
    }
    // Backward solve L^T x = z
    for (int k = KD - 1; k >= 0; --k) {
        if (tid == 0) yv[k] /= Ml[TRI(k, k)];
        __syncthreads();
        float xk = yv[k];
        int kb = rbase[k];
        for (int c = tid; c < k; c += 256) yv[c] -= Ml[kb + c] * xk;
        __syncthreads();
    }
    if (tid < KD) out[rowOff + tid] = yv[tid];
}

// ---------------------------------------------------------------------------
extern "C" void kernel_launch(void* const* d_in, const int* in_sizes, int n_in,
                              void* d_out, int out_size, void* d_ws, size_t ws_size,
                              hipStream_t stream)
{
    (void)in_sizes; (void)n_in; (void)out_size; (void)ws_size;
    const float* feat_x  = (const float*)d_in[0];
    const float* feat_y  = (const float*)d_in[1];
    const float* evals_x = (const float*)d_in[2];
    const float* evals_y = (const float*)d_in[3];
    const float* evtx    = (const float*)d_in[4];
    const float* evty    = (const float*)d_in[5];
    float* out = (float*)d_out;
    float* ws  = (float*)d_ws;

    const size_t nPart = (size_t)SPLITK * NB * KD * CD;  // 2,097,152
    const size_t nProj = (size_t)NB * KD * CD;           //   262,144
    const size_t nKK   = (size_t)NB * KD * KD;           //   131,072
    float* partX = ws;                 // 8.39 MB
    float* partY = partX + nPart;      // 8.39 MB
    float* A     = partY + nPart;      // 1 MB
    float* Bm    = A + nProj;          // 1 MB
    float* AAt   = Bm + nProj;         // 0.5 MB
    float* BAt   = AAt + nKK;          // 0.5 MB
    float* Dm    = BAt + nKK;          // 0.5 MB  (total ~19.5 MB of ws)

    dim3 gGemm(SPLITK, CD / TC, NB);   // 8 x 4 x 8 = 256 workgroups
    proj_gemm_kernel<<<gGemm, 256, 0, stream>>>(evtx, feat_x, partX);
    proj_gemm_kernel<<<gGemm, 256, 0, stream>>>(evty, feat_y, partY);

    const int nRed = (int)nProj;
    reduce_kernel<<<(nRed + 255) / 256, 256, 0, stream>>>(partX, A, nRed);
    reduce_kernel<<<(nRed + 255) / 256, 256, 0, stream>>>(partY, Bm, nRed);

    gram_kernel<<<dim3(4, 4, NB * 2), 256, 0, stream>>>(A, Bm, AAt, BAt);
    mask_kernel<<<NB, 256, 0, stream>>>(evals_x, evals_y, Dm);
    solve_kernel<<<dim3(KD, NB), 256, 0, stream>>>(AAt, BAt, Dm, out);
}
